// LSTMModel_23828478558740
// MI455X (gfx1250) — compile-verified, matrix operand order checked by reading, not run on hardware
//
#include <hip/hip_runtime.h>
#include <hip/hip_fp16.h>

#define HDIM   100
#define GDIM   400    // 4*H (gate rows: i, f, g, o)
#define TBLOCK 416    // 13 wave32s on one WGP
#define CH     32     // x-prefetch chunk (one wave32, one async b32 per chunk)

typedef __attribute__((ext_vector_type(2)))  float    v2f;
typedef __attribute__((ext_vector_type(4)))  float    v4f;
typedef __attribute__((ext_vector_type(8)))  float    v8f;
typedef __attribute__((ext_vector_type(16))) _Float16 v16h;

__device__ __forceinline__ float fast_sigmoid(float x) {
  // 1 / (1 + exp(-x)) via native v_exp_f32 (exp2) + v_rcp_f32
  float e = __builtin_amdgcn_exp2f(-1.442695040888963f * x);
  return __builtin_amdgcn_rcpf(1.0f + e);
}
__device__ __forceinline__ float fast_tanh(float x) {
  // tanh(x) = 2/(1+exp(-2x)) - 1
  float e = __builtin_amdgcn_exp2f(-2.885390081777927f * x);
  return __builtin_amdgcn_rcpf(1.0f + e) * 2.0f - 1.0f;
}

__device__ __forceinline__ void async_load_x_b32(const float* gaddr, const float* lds_ptr) {
  // GLOBAL_LOAD_ASYNC_TO_LDS_B32, GV mode: per-lane 4B memory -> LDS (ASYNCcnt)
  unsigned lds_a = (unsigned)(size_t)lds_ptr;  // low 32 bits of generic ptr = LDS byte offset
  asm volatile("global_load_async_to_lds_b32 %0, %1, off"
               :: "v"(lds_a), "v"(gaddr) : "memory");
}
__device__ __forceinline__ void wait_async0() {
#if __has_builtin(__builtin_amdgcn_s_wait_asynccnt)
  __builtin_amdgcn_s_wait_asynccnt(0);
#else
  asm volatile("s_wait_asynccnt 0x0" ::: "memory");
#endif
}

// Packed dual-FMA: acc.{lo,hi} += a.{lo,hi} * b.{lo,hi}  (V_PK_FMA_F32, VOP3P)
__device__ __forceinline__ void pk_fma(v2f& acc, v2f a, v2f b) {
  asm("v_pk_fma_f32 %0, %1, %2, %0" : "+v"(acc) : "v"(a), "v"(b));
}

__global__ __launch_bounds__(TBLOCK)
void lstm_persistent(const float* __restrict__ x,     // (T)
                     const float* __restrict__ W_ih,  // (4H, 1)
                     const float* __restrict__ W_hh,  // (4H, H) row-major
                     const float* __restrict__ b_ih,  // (4H)
                     const float* __restrict__ b_hh,  // (4H)
                     const float* __restrict__ W_lin, // (1, H)
                     const float* __restrict__ b_lin, // (1)
                     float* __restrict__ out, int T)
{
  __shared__ __align__(16) float hbuf[HDIM];     // h_{t-1}, broadcast to all waves
  __shared__ float gbuf[GDIM];                   // pre-activation gates for step t
  __shared__ __align__(16) float xbuf[2][CH];    // double-buffered x stream

  const int tid = threadIdx.x;

  // ---- one-time setup: W_hh row resident in VGPRs as 50 packed f32 pairs ----
  v2f w2[HDIM / 2];
  float xw = 0.f, bj = 0.f;
  if (tid < GDIM) {
    const float4* wr = (const float4*)(W_hh + tid * HDIM);  // 400B rows -> 16B aligned
#pragma unroll
    for (int k = 0; k < HDIM / 4; ++k) {
      float4 v = wr[k];
      w2[2*k+0][0] = v.x; w2[2*k+0][1] = v.y;
      w2[2*k+1][0] = v.z; w2[2*k+1][1] = v.w;
    }
    xw = W_ih[tid];
    bj = b_ih[tid] + b_hh[tid];
  }
  if (tid < HDIM) hbuf[tid] = 0.f;
  float cst = 0.f;  // cell state, owned by threads 0..99

  // ---- prime async x pipeline: chunk 0 -> xbuf[0] ----
  if (tid < CH) {
    if (tid < T) async_load_x_b32(x + tid, &xbuf[0][tid]);
  }

  // ---- sequential recurrence ----
  for (int t = 0; t < T; ++t) {
    const int c = t >> 5;  // chunk index
    if ((t & (CH - 1)) == 0) {
      if (tid < CH) {
        wait_async0();                       // chunk c resident in xbuf[c&1]
        int gidx = (c + 1) * CH + tid;       // issue chunk c+1 -> xbuf[(c+1)&1]
        if (gidx < T) async_load_x_b32(x + gidx, &xbuf[(c + 1) & 1][tid]);
      }
    }
    __syncthreads();  // publish h_{t-1} (and fresh x chunk) to all waves

    const float x_t = xbuf[c & 1][t & (CH - 1)];

    if (tid < GDIM) {
      // g[j] = x_t*W_ih[j] + b_ih[j]+b_hh[j] + dot(W_hh[j,:], h)
      v2f acc0 = {fmaf(x_t, xw, bj), 0.f};
      v2f acc1 = {0.f, 0.f};
      const v4f* h4 = (const v4f*)hbuf;
#pragma unroll
      for (int k = 0; k < HDIM / 4; ++k) {
        v4f hv = h4[k];   // ds_load_b128, same-address broadcast across lanes
        v2f h01 = __builtin_shufflevector(hv, hv, 0, 1);
        v2f h23 = __builtin_shufflevector(hv, hv, 2, 3);
        pk_fma(acc0, w2[2*k+0], h01);   // v_pk_fma_f32: 2 FMAs / issue
        pk_fma(acc1, w2[2*k+1], h23);
      }
      gbuf[tid] = (acc0[0] + acc0[1]) + (acc1[0] + acc1[1]);
    }
    __syncthreads();  // gates complete

    if (tid < HDIM) {
      float gi = gbuf[tid];
      float gf = gbuf[tid + HDIM];
      float gg = gbuf[tid + 2 * HDIM];
      float go = gbuf[tid + 3 * HDIM];
      cst = fast_sigmoid(gf) * cst + fast_sigmoid(gi) * fast_tanh(gg);
      hbuf[tid] = fast_sigmoid(go) * fast_tanh(cst);
    }
  }
  __syncthreads();

  // ---- final projection: out = h . W_lin + b_lin, via WMMA (wave 0, EXEC all-ones) ----
  if (tid < 32) {
    const int lane = tid;
    v8f acc = {0.f, 0.f, 0.f, 0.f, 0.f, 0.f, 0.f, 0.f};
#if __has_builtin(__builtin_amdgcn_wmma_f32_16x16x4_f32)
    // A(16x4 f32): lanes 0-15 hold A[m][K0],A[m][K1]; lanes 16-31 hold A[m][K2],A[m][K3].
    // Row 0 of A = h chunk; column 0 of B = W_lin chunk; D[0][0] (lane 0, elem 0) = dot.
#pragma unroll
    for (int kc = 0; kc < HDIM / 4; ++kc) {
      v2f A = {0.f, 0.f}, B = {0.f, 0.f};
      if (lane == 0)  { A[0] = hbuf[4*kc+0];  A[1] = hbuf[4*kc+1];
                        B[0] = W_lin[4*kc+0]; B[1] = W_lin[4*kc+1]; }
      if (lane == 16) { A[0] = hbuf[4*kc+2];  A[1] = hbuf[4*kc+3];
                        B[0] = W_lin[4*kc+2]; B[1] = W_lin[4*kc+3]; }
      acc = __builtin_amdgcn_wmma_f32_16x16x4_f32(false, A, false, B,
                                                  (short)0, acc, false, false);
    }
#else
    // Fallback: codegen-confirmed f16 WMMA, K padded 100 -> 128 over 4 chunks of 32.
#pragma unroll
    for (int kc = 0; kc < 4; ++kc) {
      v16h A, B;
#pragma unroll
      for (int e = 0; e < 16; ++e) { A[e] = (_Float16)0.f; B[e] = (_Float16)0.f; }
      if (lane == 0 || lane == 16) {
        int base = kc * 32 + (lane == 0 ? 0 : 8);
#pragma unroll
        for (int e = 0; e < 8; ++e) {
          int k0 = base + e, k1 = base + 16 + e;
          if (k0 < HDIM) { A[e]     = (_Float16)hbuf[k0];  B[e]     = (_Float16)W_lin[k0]; }
          if (k1 < HDIM) { A[e + 8] = (_Float16)hbuf[k1];  B[e + 8] = (_Float16)W_lin[k1]; }
        }
      }
      acc = __builtin_amdgcn_wmma_f32_16x16x32_f16(false, A, false, B,
                                                   (short)0, acc, false, false);
    }
#endif
    if (lane == 0) out[0] = acc[0] + b_lin[0];
  }
}

extern "C" void kernel_launch(void* const* d_in, const int* in_sizes, int n_in,
                              void* d_out, int out_size, void* d_ws, size_t ws_size,
                              hipStream_t stream) {
  (void)n_in; (void)out_size; (void)d_ws; (void)ws_size;
  const float* x     = (const float*)d_in[0];  // (1, T, 1)
  const float* W_ih  = (const float*)d_in[1];  // (4H, 1)
  const float* W_hh  = (const float*)d_in[2];  // (4H, H)
  const float* b_ih  = (const float*)d_in[3];  // (4H)
  const float* b_hh  = (const float*)d_in[4];  // (4H)
  const float* W_lin = (const float*)d_in[5];  // (1, H)
  const float* b_lin = (const float*)d_in[6];  // (1)
  const int T = in_sizes[0];                   // 200000

  lstm_persistent<<<1, TBLOCK, 0, stream>>>(x, W_ih, W_hh, b_ih, b_hh,
                                            W_lin, b_lin, (float*)d_out, T);
}
